// KNNAttention_80582176407767
// MI455X (gfx1250) — matrix operand order repaired and checked by reading, not compiled, not run on hardware
//
#include <hip/hip_runtime.h>
#include <hip/hip_bf16.h>

typedef __attribute__((ext_vector_type(16))) __bf16 v16bf;
typedef __attribute__((ext_vector_type(8)))  float  v8f;
typedef __bf16 bf16_t;

#define Bc   2
#define Nc   2048
#define DIMc 512
#define Hc   8
#define DHc  64
#define Kc   32
#define SCALEc 0.125f   // DH^-0.5

// ---------------------------------------------------------------------------
// Fragment loaders: force global_load_b128 / ds_load_b128 instead of u16 soup.
// ---------------------------------------------------------------------------
__device__ __forceinline__ v16bf load_frag_b(const bf16_t* p) {
  // 16 contiguous bf16 (32B, 16B-aligned): two b128 loads
  union { v16bf v; uint4 u[2]; } t;
  t.u[0] = *(const uint4*)(p);
  t.u[1] = *(const uint4*)(p + 8);
  return t.v;
}
__device__ __forceinline__ v16bf load_frag_a(const bf16_t* p) {
  // A 16-bit frag: 16 contiguous elements (bytes +0 / +16)
  union { v16bf v; uint4 u[2]; } t;
  t.u[0] = *(const uint4*)(p);
  t.u[1] = *(const uint4*)(p + 8);
  return t.v;
}

// ---------------------------------------------------------------------------
// f32 -> bf16 convert (n divisible by 4)
// ---------------------------------------------------------------------------
__global__ void __launch_bounds__(256) cvt_f32_bf16_kernel(
    const float* __restrict__ src, bf16_t* __restrict__ dst, int n) {
  int i = (blockIdx.x * blockDim.x + threadIdx.x) * 4;
  if (i < n) {
    float4 v = *(const float4*)(src + i);
    dst[i + 0] = (bf16_t)v.x;
    dst[i + 1] = (bf16_t)v.y;
    dst[i + 2] = (bf16_t)v.z;
    dst[i + 3] = (bf16_t)v.w;
  }
}

// ---------------------------------------------------------------------------
// f32 W[rows,cols] -> bf16 Wt[cols,rows] (transpose for contiguous B-frags)
// ---------------------------------------------------------------------------
__global__ void __launch_bounds__(256) cvt_transpose_bf16_kernel(
    const float* __restrict__ W, bf16_t* __restrict__ Wt, int rows, int cols) {
  int idx = blockIdx.x * blockDim.x + threadIdx.x;
  if (idx >= rows * cols) return;
  int k = idx % rows;              // output-fastest: coalesced Wt writes
  int n = idx / rows;
  Wt[(size_t)n * rows + k] = (bf16_t)W[(size_t)k * cols + n];
}

// ---------------------------------------------------------------------------
// kv f32 [B*N, 128] -> k bf16 [B*N, 64]  (row-major, S B-frags contiguous)
// ---------------------------------------------------------------------------
__global__ void __launch_bounds__(256) pack_k_kernel(
    const float* __restrict__ kvf, bf16_t* __restrict__ kb) {
  int idx = blockIdx.x * blockDim.x + threadIdx.x;   // (b*N+j)*64 + d
  int d = idx % DHc;
  int r = idx / DHc;                                 // b*N + j
  kb[idx] = (bf16_t)kvf[(size_t)r * (2 * DHc) + d];
}

// ---------------------------------------------------------------------------
// kv f32 [B*N, 128] -> vT bf16 [B, DH, N]  (P@V B-frags contiguous along j)
// ---------------------------------------------------------------------------
__global__ void __launch_bounds__(256) pack_vT_kernel(
    const float* __restrict__ kvf, bf16_t* __restrict__ vT) {
  int idx = blockIdx.x * blockDim.x + threadIdx.x;   // (b*DH + d)*N + j
  int j = idx % Nc;
  int d = (idx / Nc) % DHc;
  int b = idx / (Nc * DHc);
  vT[idx] = (bf16_t)kvf[((size_t)(b * Nc + j)) * (2 * DHc) + DHc + d];
}

// ---------------------------------------------------------------------------
// bf16 WMMA GEMM: C[M,Nn] f32 = A[M,Kk] @ BT[Nn,Kk]^T (+ bias[Nn]).
// 128 threads = 4 waves; one 16x16 C tile per wave; all frag loads are b128.
// ---------------------------------------------------------------------------
__global__ void __launch_bounds__(128) gemm_bf16_wmma_kernel(
    const bf16_t* __restrict__ A, const bf16_t* __restrict__ BT,
    const float* __restrict__ bias, float* __restrict__ C,
    int M, int Nn, int Kk) {
  const int lane = threadIdx.x & 31;
  const int wave = threadIdx.x >> 5;
  const int m0 = blockIdx.x * 16;
  const int n0 = (blockIdx.y * 4 + wave) * 16;
  if (n0 >= Nn) return;                  // wave-uniform: EXEC stays all-ones

  const int hi = lane >> 4;
  const int lo = lane & 15;
  const bf16_t* arow = A  + (size_t)(m0 + lo) * Kk + hi * 16;  // A frag base
  const bf16_t* brow = BT + (size_t)(n0 + lo) * Kk + hi * 16;  // B frag base

  v8f acc = {};
  for (int kk = 0; kk < Kk; kk += 32) {
    v16bf a = load_frag_a(arow + kk);
    v16bf b = load_frag_b(brow + kk);
    acc = __builtin_amdgcn_wmma_f32_16x16x32_bf16(false, a, false, b,
                                                  (short)0, acc, false, false);
  }
  const float bv = bias ? bias[n0 + lo] : 0.0f;
#pragma unroll
  for (int r = 0; r < 8; ++r)
    C[(size_t)(m0 + hi * 8 + r) * Nn + n0 + lo] = acc[r] + bv;
}

// ---------------------------------------------------------------------------
// Causal local attention, flash style. One wave per (b, h, 16-row q tile).
// Key tiles of 32; S via 2 WMMAs (DH=64=2xK32); online softmax; P restaged
// through wave-private LDS into A-frag layout; O += P@V via 4 WMMAs.
//   qb : [B*N, H*DH] bf16, kb : [B*N, DH] bf16, vT : [B, DH, N] bf16
// ---------------------------------------------------------------------------
__global__ void __launch_bounds__(32) flash_local_kernel(
    const bf16_t* __restrict__ qb16, const bf16_t* __restrict__ kb16,
    const bf16_t* __restrict__ vT16, float* __restrict__ outLocal) {
  __shared__ __align__(16) float sP[16 * 32];
  const int lane = threadIdx.x;
  const int i0 = blockIdx.x * 16;
  const int h  = blockIdx.y;
  const int b  = blockIdx.z;
  const int hi = lane >> 4;
  const int lo = lane & 15;

  const bf16_t* kbb = kb16 + (size_t)b * Nc * DHc;
  const bf16_t* vTb = vT16 + (size_t)b * DHc * Nc;

  // Q A-fragments: 16x64 -> two 16x32 bf16 frags (b128 loads)
  const bf16_t* qrow =
      qb16 + (size_t)(b * Nc + i0 + lo) * (Hc * DHc) + h * DHc + hi * 16;
  const v16bf qa0 = load_frag_a(qrow);
  const v16bf qa1 = load_frag_a(qrow + 32);

  float rowmax[8], rowsum[8];
#pragma unroll
  for (int r = 0; r < 8; ++r) { rowmax[r] = -3.0e38f; rowsum[r] = 0.0f; }
  v8f o[4] = {{}, {}, {}, {}};

  const int jend = i0 + 15;
  for (int j0 = 0; j0 <= jend; j0 += 32) {
    // ---- S = (Q K^T) * scale : 16x32 tile as two 16x16 C frags ----
    float s[2][8];
#pragma unroll
    for (int t = 0; t < 2; ++t) {
      const bf16_t* krow = kbb + (size_t)(j0 + t * 16 + lo) * DHc + hi * 16;
      v8f st = {};
      st = __builtin_amdgcn_wmma_f32_16x16x32_bf16(false, qa0, false,
              load_frag_b(krow),      (short)0, st, false, false);
      st = __builtin_amdgcn_wmma_f32_16x16x32_bf16(false, qa1, false,
              load_frag_b(krow + 32), (short)0, st, false, false);
#pragma unroll
      for (int r = 0; r < 8; ++r) {
        float sv = st[r] * SCALEc;
        if (j0 + t * 16 + lo > i0 + hi * 8 + r) sv = -3.0e38f;  // causal
        s[t][r] = sv;
      }
    }

    // ---- online softmax: row stats at (vgpr r, lane-half hi) ----
    float alpha[8];
#pragma unroll
    for (int r = 0; r < 8; ++r) {
      float tm = fmaxf(s[0][r], s[1][r]);
#pragma unroll
      for (int off = 1; off < 16; off <<= 1)
        tm = fmaxf(tm, __shfl_xor(tm, off, 32));
      const float mnew = fmaxf(rowmax[r], tm);
      alpha[r] = __expf(rowmax[r] - mnew);
      rowmax[r] = mnew;
      const float p0 = __expf(s[0][r] - mnew);
      const float p1 = __expf(s[1][r] - mnew);
      sP[(hi * 8 + r) * 32 + lo]      = p0;
      sP[(hi * 8 + r) * 32 + 16 + lo] = p1;
      float ps = p0 + p1;
#pragma unroll
      for (int off = 1; off < 16; off <<= 1)
        ps += __shfl_xor(ps, off, 32);
      rowsum[r] = rowsum[r] * alpha[r] + ps;
    }
    __syncthreads();   // single-wave WG -> S_NOP; LDS in-order per wave

#pragma unroll
    for (int t = 0; t < 4; ++t)
#pragma unroll
      for (int r = 0; r < 8; ++r) o[t][r] *= alpha[r];

    // ---- P as 16x32 bf16 A-frag from LDS (ds_load_b128) ----
    v16bf pa;
    {
      const float* pp = &sP[lo * 32 + hi * 16];
      const float4 c0 = *(const float4*)(pp);
      const float4 c1 = *(const float4*)(pp + 4);
      const float4 c2 = *(const float4*)(pp + 8);
      const float4 c3 = *(const float4*)(pp + 12);
      pa[0] = (__bf16)c0.x; pa[1] = (__bf16)c0.y;
      pa[2] = (__bf16)c0.z; pa[3] = (__bf16)c0.w;
      pa[4] = (__bf16)c1.x; pa[5] = (__bf16)c1.y;
      pa[6] = (__bf16)c1.z; pa[7] = (__bf16)c1.w;
      pa[8]  = (__bf16)c2.x; pa[9]  = (__bf16)c2.y;
      pa[10] = (__bf16)c2.z; pa[11] = (__bf16)c2.w;
      pa[12] = (__bf16)c3.x; pa[13] = (__bf16)c3.y;
      pa[14] = (__bf16)c3.z; pa[15] = (__bf16)c3.w;
    }

    // ---- O += P @ V : V from transposed copy, contiguous b128 frags ----
#pragma unroll
    for (int t = 0; t < 4; ++t) {
      const bf16_t* vrow = vTb + (size_t)(t * 16 + lo) * Nc + j0 + hi * 16;
      o[t] = __builtin_amdgcn_wmma_f32_16x16x32_bf16(false, pa, false,
                load_frag_b(vrow), (short)0, o[t], false, false);
    }
    __syncthreads();
  }

  float* ob = outLocal + (size_t)b * Nc * (Hc * DHc) + h * DHc;
#pragma unroll
  for (int r = 0; r < 8; ++r) {
    const float inv = 1.0f / rowsum[r];
    const size_t row = (size_t)(i0 + hi * 8 + r) * (Hc * DHc);
#pragma unroll
    for (int t = 0; t < 4; ++t)
      ob[row + t * 16 + lo] = o[t][r] * inv;
  }
}

// ---------------------------------------------------------------------------
// Memory attention: one wave per (b,h,i); lane = memory slot. Streams the
// 0.54 GB mem_kv tensor -> this is the roofline-limiting kernel (~23 us at
// 23.3 TB/s). float4 loads, prefetch, wave32 butterfly reductions.
// ---------------------------------------------------------------------------
__global__ void __launch_bounds__(256) mem_attn_kernel(
    const float* __restrict__ q, const float* __restrict__ mem_kv,
    const unsigned char* __restrict__ mem_mask,
    const float* __restrict__ null_k, const float* __restrict__ null_v,
    float* __restrict__ outMem) {
  const int lane = threadIdx.x & 31;
  const int wid  = blockIdx.x * (blockDim.x >> 5) + (threadIdx.x >> 5);
  const int i = wid % Nc;
  const int h = (wid / Nc) % Hc;
  const int b = wid / (Nc * Hc);

  const float* qrow = q + (size_t)(b * Nc + i) * (Hc * DHc) + h * DHc;
  const float* krow = mem_kv +
      ((((size_t)(b * Hc + h) * Nc + i) * Kc + lane) * 2) * DHc;
  const float* vrow = krow + DHc;
  __builtin_prefetch(vrow, 0, 1);   // global_prefetch_b8: warm v while dotting k

  float s = 0.0f, sn = 0.0f;
#pragma unroll
  for (int d = 0; d < DHc; d += 4) {
    const float4 qd = *(const float4*)(qrow + d);
    const float4 kd = *(const float4*)(krow + d);
    const float4 nk = *(const float4*)(null_k + d);
    s  += qd.x * kd.x + qd.y * kd.y + qd.z * kd.z + qd.w * kd.w;
    sn += qd.x * nk.x + qd.y * nk.y + qd.z * nk.z + qd.w * nk.w;
  }
  s *= SCALEc; sn *= SCALEc;
  const bool ok =
      mem_mask[((size_t)(b * Hc + h) * Nc + i) * Kc + lane] != 0;
  if (!ok) s = -3.0e38f;

  float m = fmaxf(s, sn);
#pragma unroll
  for (int off = 1; off < 32; off <<= 1) m = fmaxf(m, __shfl_xor(m, off, 32));
  const float w  = __expf(s - m);    // 0 when masked
  const float wn = __expf(sn - m);   // identical on all lanes
  float denom = w;
#pragma unroll
  for (int off = 1; off < 32; off <<= 1) denom += __shfl_xor(denom, off, 32);
  denom += wn;
  const float inv = 1.0f / denom;

  float* orow = outMem + (size_t)(b * Nc + i) * (Hc * DHc) + h * DHc;
#pragma unroll
  for (int d0 = 0; d0 < DHc; d0 += 4) {
    const float4 vv = *(const float4*)(vrow + d0);
    float ax = w * vv.x, ay = w * vv.y, az = w * vv.z, aw = w * vv.w;
#pragma unroll
    for (int off = 1; off < 32; off <<= 1) {
      ax += __shfl_xor(ax, off, 32);
      ay += __shfl_xor(ay, off, 32);
      az += __shfl_xor(az, off, 32);
      aw += __shfl_xor(aw, off, 32);
    }
    if (lane == 0) {
      const float4 nv = *(const float4*)(null_v + d0);
      float4 res;
      res.x = (ax + wn * nv.x) * inv;
      res.y = (ay + wn * nv.y) * inv;
      res.z = (az + wn * nv.z) * inv;
      res.w = (aw + wn * nv.w) * inv;
      *(float4*)(orow + d0) = res;
    }
  }
}

// ---------------------------------------------------------------------------
// Gated combine -> bf16 for the output projection.
// ---------------------------------------------------------------------------
__global__ void __launch_bounds__(256) combine_gate_kernel(
    const float* __restrict__ localv, const float* __restrict__ memv,
    const float* __restrict__ gate, bf16_t* __restrict__ y) {
  const size_t idx = (size_t)blockIdx.x * blockDim.x + threadIdx.x;
  const int hh = ((int)(idx % (Hc * DHc))) / DHc;
  const float g = 1.0f / (1.0f + __expf(-gate[hh]));
  y[idx] = (bf16_t)(localv[idx] * g + memv[idx] * (1.0f - g));
}

// ---------------------------------------------------------------------------
// Orchestration
// ---------------------------------------------------------------------------
extern "C" void kernel_launch(void* const* d_in, const int* in_sizes, int n_in,
                              void* d_out, int out_size, void* d_ws, size_t ws_size,
                              hipStream_t stream) {
  const float*         x        = (const float*)d_in[0];
  const float*         mem_kv   = (const float*)d_in[1];
  const unsigned char* mem_mask = (const unsigned char*)d_in[2];
  const float*         Wq       = (const float*)d_in[3];
  const float*         Wkv      = (const float*)d_in[4];
  const float*         Wo       = (const float*)d_in[5];
  const float*         bo       = (const float*)d_in[6];
  const float*         null_k   = (const float*)d_in[7];
  const float*         null_v   = (const float*)d_in[8];
  const float*         gate     = (const float*)d_in[9];
  float* out = (float*)d_out;

  const int BN = Bc * Nc;                       // 4096
  char* ws = (char*)d_ws;
  size_t off = 0;
  auto alloc = [&](size_t bytes) {
    void* p = ws + off;
    off += (bytes + 255) & ~(size_t)255;
    return p;
  };
  bf16_t* xb    = (bf16_t*)alloc((size_t)BN * DIMc * 2);
  bf16_t* WqT   = (bf16_t*)alloc((size_t)DIMc * (Hc * DHc) * 2);   // [N=512][K=512]
  bf16_t* WkvT  = (bf16_t*)alloc((size_t)DIMc * (2 * DHc) * 2);    // [N=128][K=512]
  bf16_t* WoT   = (bf16_t*)alloc((size_t)(Hc * DHc) * DIMc * 2);   // [N=512][K=512]
  float*  qf    = (float*) alloc((size_t)BN * (Hc * DHc) * 4);
  float*  kvf   = (float*) alloc((size_t)BN * (2 * DHc) * 4);
  bf16_t* qb16  = (bf16_t*)alloc((size_t)BN * (Hc * DHc) * 2);
  bf16_t* kb16  = (bf16_t*)alloc((size_t)BN * DHc * 2);
  bf16_t* vT16  = (bf16_t*)alloc((size_t)Bc * DHc * Nc * 2);
  float*  locf  = (float*) alloc((size_t)BN * (Hc * DHc) * 4);
  float*  memf  = (float*) alloc((size_t)BN * (Hc * DHc) * 4);
  bf16_t* yb    = (bf16_t*)alloc((size_t)BN * (Hc * DHc) * 2);
  (void)ws_size; (void)in_sizes; (void)n_in; (void)out_size;

  // 1) converts / weight transposes
  {
    int n = BN * DIMc;
    cvt_f32_bf16_kernel<<<n / 4 / 256, 256, 0, stream>>>(x, xb, n);
    n = DIMc * (Hc * DHc);
    cvt_transpose_bf16_kernel<<<(n + 255) / 256, 256, 0, stream>>>(
        Wq, WqT, DIMc, Hc * DHc);
    n = DIMc * (2 * DHc);
    cvt_transpose_bf16_kernel<<<(n + 255) / 256, 256, 0, stream>>>(
        Wkv, WkvT, DIMc, 2 * DHc);
    n = (Hc * DHc) * DIMc;
    cvt_transpose_bf16_kernel<<<(n + 255) / 256, 256, 0, stream>>>(
        Wo, WoT, Hc * DHc, DIMc);
  }

  // 2) q = x@Wq  [4096x512x512], kv = x@Wkv  [4096x512x128]
  gemm_bf16_wmma_kernel<<<dim3(BN / 16, (Hc * DHc) / 64), 128, 0, stream>>>(
      xb, WqT, nullptr, qf, BN, Hc * DHc, DIMc);
  gemm_bf16_wmma_kernel<<<dim3(BN / 16, (2 * DHc) / 64), 128, 0, stream>>>(
      xb, WkvT, nullptr, kvf, BN, 2 * DHc, DIMc);

  // 3) pack attention operands to bf16 fragment-friendly layouts
  {
    int n = BN * (Hc * DHc);
    cvt_f32_bf16_kernel<<<n / 4 / 256, 256, 0, stream>>>(qf, qb16, n);
    n = BN * DHc;
    pack_k_kernel<<<n / 256, 256, 0, stream>>>(kvf, kb16);
    pack_vT_kernel<<<n / 256, 256, 0, stream>>>(kvf, vT16);
  }

  // 4) causal local attention (flash, WMMA)
  flash_local_kernel<<<dim3(Nc / 16, Hc, Bc), 32, 0, stream>>>(
      qb16, kb16, vT16, locf);

  // 5) memory attention (roofline-limiting mem_kv stream)
  mem_attn_kernel<<<(Bc * Hc * Nc) / 8, 256, 0, stream>>>(
      qf, mem_kv, mem_mask, null_k, null_v, memf);

  // 6) gated combine -> bf16
  combine_gate_kernel<<<(BN * (Hc * DHc)) / 256, 256, 0, stream>>>(
      locf, memf, gate, yb);

  // 7) out = y@Wo + bo  [4096x512x512]
  gemm_bf16_wmma_kernel<<<dim3(BN / 16, DIMc / 64), 128, 0, stream>>>(
      yb, WoT, bo, out, BN, DIMc, DIMc);
}
